// DeformableTransformerDecoderLayer_24507083391015
// MI455X (gfx1250) — matrix operand order repaired and checked by reading, not compiled, not run on hardware
//
#include <hip/hip_runtime.h>
#include <hip/hip_bf16.h>
#include <math.h>

typedef __attribute__((ext_vector_type(16))) _Float16 v16h;
typedef __attribute__((ext_vector_type(8)))  float    v8f;
typedef __attribute__((ext_vector_type(4)))  float    f32x4;

#define B_    8
#define Q_    900
#define D_    256
#define H_    8
#define L_    4
#define P_    4
#define DH_   32
#define DFF_  1024
#define S_    21760
#define NQ_   (B_*Q_)

// ---------------------------------------------------------------------------
// WMMA fragment helpers (wave32, V_WMMA_F32_16X16X32_F16)
//   A 16x32 f16 layout: lane(0..15)=row M; elems {half*8+j, 16+half*8+j}
//   B 32x16 f16 layout: lane(0..15)=col N; elems {half*8+j, 16+half*8+j}
//   C 16x16 f32 layout: lane=col N; vgpr i = row half*8+i
// ---------------------------------------------------------------------------
__device__ inline v16h load_a_f32(const float* __restrict__ p, int half) {
    v16h f;
#pragma unroll
    for (int j = 0; j < 8; ++j) f[j]     = (_Float16)p[half * 8 + j];
#pragma unroll
    for (int j = 0; j < 8; ++j) f[8 + j] = (_Float16)p[16 + half * 8 + j];
    return f;
}

__device__ inline v16h load_b_f32(const float* __restrict__ p, int half) {
    v16h f;
#pragma unroll
    for (int j = 0; j < 16; ++j) f[j] = (_Float16)p[(j & 8) * 2 + half * 8 + (j & 7)];
    return f;
}

__device__ inline v16h cvt16(const f32x4 r[4]) {
    v16h f;
#pragma unroll
    for (int q = 0; q < 4; ++q)
#pragma unroll
        for (int j = 0; j < 4; ++j) f[q * 4 + j] = (_Float16)r[q][j];
    return f;
}

// ---------------------------------------------------------------------------
// Generic GEMM:  C[M,N] = A[M,K] @ W[N,K]^T + bias  (optional ReLU)
// block = 256 threads = 8 waves arranged 2(M) x 4(N); wave tile 16 x 64.
// grid = (ceil(M/32), ceil(N/256))
// Branchless, software-pipelined K-loop: all 20 b128 loads of iteration k+1
// are issued (into distinct registers) before the 4 WMMAs of iteration k, so
// loads stay in flight under the matrix ops instead of serializing on
// s_wait_loadcnt 0.  Out-of-range rows/cols are clamped (garbage lands only
// in store-guarded elements).
// ---------------------------------------------------------------------------
__global__ __launch_bounds__(256) void gemm_wmma_nt(
    const float* __restrict__ A, const float* __restrict__ W,
    const float* __restrict__ bias, float* __restrict__ C,
    int M, int N, int K, int relu)
{
    const int lane = threadIdx.x & 31;
    const int wave = threadIdx.x >> 5;
    const int wm = wave >> 2, wn = wave & 3;
    const int half = lane >> 4, l16 = lane & 15;
    const int row0 = blockIdx.x * 32 + wm * 16;
    const int col0 = blockIdx.y * 256 + wn * 64;

    v8f acc[4];
#pragma unroll
    for (int t = 0; t < 4; ++t)
#pragma unroll
        for (int i = 0; i < 8; ++i) acc[t][i] = 0.0f;

    const int arow = min(row0 + l16, M - 1);
    const float* arow_p = A + (size_t)arow * K;

    const float* brow_p[4];
#pragma unroll
    for (int t = 0; t < 4; ++t) {
        int bcol = min(col0 + t * 16 + l16, N - 1);
        brow_p[t] = W + (size_t)bcol * K;
    }

    f32x4 araw[4];
    f32x4 braw[4][4];

    auto load_a_raw = [&](int k0, f32x4 r[4]) {
        const f32x4* ap = (const f32x4*)(arow_p + k0);
        r[0] = ap[half * 2];
        r[1] = ap[half * 2 + 1];
        r[2] = ap[4 + half * 2];
        r[3] = ap[4 + half * 2 + 1];
    };
    auto load_b_raw = [&](int k0, int t, f32x4 r[4]) {
        const f32x4* bp = (const f32x4*)(brow_p[t] + k0);
        r[0] = bp[half * 2];
        r[1] = bp[half * 2 + 1];
        r[2] = bp[4 + half * 2];
        r[3] = bp[4 + half * 2 + 1];
    };

    // prologue: loads for k0 = 0
    load_a_raw(0, araw);
#pragma unroll
    for (int t = 0; t < 4; ++t) load_b_raw(0, t, braw[t]);

    for (int k0 = 0; k0 < K; k0 += 32) {
        // convert current iteration's raw data to f16 fragments
        v16h af  = cvt16(araw);
        v16h bf0 = cvt16(braw[0]);
        v16h bf1 = cvt16(braw[1]);
        v16h bf2 = cvt16(braw[2]);
        v16h bf3 = cvt16(braw[3]);

        // issue next iteration's loads (branchless: last iter reloads k0)
        int kn = (k0 + 32 < K) ? (k0 + 32) : k0;
        load_a_raw(kn, araw);
#pragma unroll
        for (int t = 0; t < 4; ++t) load_b_raw(kn, t, braw[t]);

        // matrix ops overlap the in-flight loads above
        acc[0] = __builtin_amdgcn_wmma_f32_16x16x32_f16(false, af, false, bf0, (short)0, acc[0], false, false);
        acc[1] = __builtin_amdgcn_wmma_f32_16x16x32_f16(false, af, false, bf1, (short)0, acc[1], false, false);
        acc[2] = __builtin_amdgcn_wmma_f32_16x16x32_f16(false, af, false, bf2, (short)0, acc[2], false, false);
        acc[3] = __builtin_amdgcn_wmma_f32_16x16x32_f16(false, af, false, bf3, (short)0, acc[3], false, false);
    }

#pragma unroll
    for (int t = 0; t < 4; ++t) {
        int col = col0 + t * 16 + l16;
        if (col >= N) continue;
        float bv = bias ? bias[col] : 0.0f;
#pragma unroll
        for (int i = 0; i < 8; ++i) {
            int row = row0 + half * 8 + i;
            if (row < M) {
                float v = acc[t][i] + bv;
                if (relu) v = fmaxf(v, 0.0f);
                C[(size_t)row * N + col] = v;
            }
        }
    }
}

// ---------------------------------------------------------------------------
// Flash attention: one wave per (b, h, 16-query tile). DH = 32 = one WMMA K.
// grid = (B*H, ceil(Q/16)), block = 32.
// ---------------------------------------------------------------------------
__global__ __launch_bounds__(32) void attn_flash(
    const float* __restrict__ Qp, const float* __restrict__ Kp,
    const float* __restrict__ Vp, float* __restrict__ O)
{
    __shared__ _Float16 plds[16 * 32];   // staged P tile (queries x keys)

    const int lane = threadIdx.x & 31;
    const int half = lane >> 4, l16 = lane & 15;
    const int bh = blockIdx.x;
    const int b = bh / H_, h = bh % H_;
    const int q0 = blockIdx.y * 16;

    const int qrow = min(q0 + l16, Q_ - 1);
    v16h aq = load_a_f32(Qp + ((size_t)(b * Q_ + qrow) * D_ + h * DH_), half);

    v8f o0, o1;
    float mrun[8], lrun[8];
#pragma unroll
    for (int i = 0; i < 8; ++i) { o0[i] = 0.f; o1[i] = 0.f; mrun[i] = -1e30f; lrun[i] = 0.f; }

    const float scale = 0.17677669529663689f;   // 1/sqrt(32)

    for (int kc = 0; kc < Q_; kc += 32) {
        // ---- scores: S = Q (16x32) @ K_chunk^T --------------------------------
        const int kA = min(kc + l16, Q_ - 1);        // keys 0..15 of chunk
        const int kB = min(kc + 16 + l16, Q_ - 1);   // keys 16..31 of chunk
        v16h bk0 = load_b_f32(Kp + ((size_t)(b * Q_ + kA) * D_ + h * DH_), half);
        v16h bk1 = load_b_f32(Kp + ((size_t)(b * Q_ + kB) * D_ + h * DH_), half);
        v8f s0, s1;
#pragma unroll
        for (int i = 0; i < 8; ++i) { s0[i] = 0.f; s1[i] = 0.f; }
        s0 = __builtin_amdgcn_wmma_f32_16x16x32_f16(false, aq, false, bk0, (short)0, s0, false, false);
        s1 = __builtin_amdgcn_wmma_f32_16x16x32_f16(false, aq, false, bk1, (short)0, s1, false, false);

        const bool ok0 = (kc + l16) < Q_;
        const bool ok1 = (kc + 16 + l16) < Q_;

        // ---- online softmax + stage P to LDS ---------------------------------
#pragma unroll
        for (int i = 0; i < 8; ++i) {
            float x0 = ok0 ? s0[i] * scale : -1e30f;
            float x1 = ok1 ? s1[i] * scale : -1e30f;
            float mx = fmaxf(x0, x1);
#pragma unroll
            for (int m = 1; m < 16; m <<= 1) mx = fmaxf(mx, __shfl_xor(mx, m, 32));
            float mnew  = fmaxf(mrun[i], mx);
            float alpha = __expf(mrun[i] - mnew);
            float p0 = __expf(x0 - mnew);
            float p1 = __expf(x1 - mnew);
            float rs = p0 + p1;
#pragma unroll
            for (int m = 1; m < 16; m <<= 1) rs += __shfl_xor(rs, m, 32);
            lrun[i] = lrun[i] * alpha + rs;
            mrun[i] = mnew;
            o0[i] *= alpha; o1[i] *= alpha;
            plds[(half * 8 + i) * 32 + l16]      = (_Float16)p0;
            plds[(half * 8 + i) * 32 + 16 + l16] = (_Float16)p1;
        }
        __syncthreads();

        // ---- P fragment (A layout) from LDS ----------------------------------
        v16h pf;
        {
            const _Float16* pr = plds + l16 * 32;
#pragma unroll
            for (int j = 0; j < 8; ++j) pf[j]     = pr[half * 8 + j];
#pragma unroll
            for (int j = 0; j < 8; ++j) pf[8 + j] = pr[16 + half * 8 + j];
        }

        // ---- V fragments (B layout): lane = dh column, K = key ----------------
        v16h v0, v1;
#pragma unroll
        for (int j = 0; j < 16; ++j) {
            int key = min(kc + (j & 8) * 2 + half * 8 + (j & 7), Q_ - 1);   // P==0 for padded keys
            const float* vp = Vp + ((size_t)(b * Q_ + key) * D_ + h * DH_);
            v0[j] = (_Float16)vp[l16];
            v1[j] = (_Float16)vp[16 + l16];
        }
        o0 = __builtin_amdgcn_wmma_f32_16x16x32_f16(false, pf, false, v0, (short)0, o0, false, false);
        o1 = __builtin_amdgcn_wmma_f32_16x16x32_f16(false, pf, false, v1, (short)0, o1, false, false);
        __syncthreads();
    }

#pragma unroll
    for (int i = 0; i < 8; ++i) {
        int qr = q0 + half * 8 + i;
        if (qr < Q_) {
            float inv = 1.0f / lrun[i];
            float* op = O + ((size_t)(b * Q_ + qr) * D_ + h * DH_);
            op[l16]      = o0[i] * inv;
            op[16 + l16] = o1[i] * inv;
        }
    }
}

// ---------------------------------------------------------------------------
// MS-deformable-attention sampling: one wave per (b,q,h); lane = channel.
// grid = ceil(B*Q*H/8), block = 256 (8 waves).
// ---------------------------------------------------------------------------
__global__ __launch_bounds__(256) void msda_sample(
    const float* __restrict__ val, const float* __restrict__ offb,
    const float* __restrict__ awl, const float* __restrict__ refp,
    float* __restrict__ out)
{
    const int lane = threadIdx.x & 31;
    const int wid = blockIdx.x * (blockDim.x >> 5) + (threadIdx.x >> 5);
    if (wid >= B_ * Q_ * H_) return;
    const int h  = wid % H_;
    const int bq = wid / H_;
    const int b  = bq / Q_;

    const int WT[4] = {128, 64, 32, 16};
    const int HT[4] = {128, 64, 32, 16};
    const int ST[4] = {0, 16384, 20480, 21504};

    const float* op = offb + (size_t)bq * 256 + h * 32; // (l,p,2)
    const float* ap = awl  + (size_t)bq * 128 + h * 16; // (l,p)
    const float* rp = refp + (size_t)bq * 8;            // (l,2)

    // softmax over the 16 attention-weight logits (lane-invariant)
    float e[16];
    float mx = -1e30f;
#pragma unroll
    for (int i = 0; i < 16; ++i) mx = fmaxf(mx, ap[i]);
    float s = 0.f;
#pragma unroll
    for (int i = 0; i < 16; ++i) { e[i] = __expf(ap[i] - mx); s += e[i]; }
    const float inv = 1.0f / s;

    const int dh = lane;
    float acc = 0.f;
#pragma unroll
    for (int l = 0; l < 4; ++l) {
        const int Wl = WT[l], Hl = HT[l];
        const float* vbase = val + ((size_t)(b * S_ + ST[l])) * 256 + h * 32 + dh;
        const float rx = rp[l * 2 + 0], ry = rp[l * 2 + 1];
#pragma unroll
        for (int p = 0; p < 4; ++p) {
            float x = rx * (float)Wl + op[l * 8 + p * 2 + 0] - 0.5f;
            float y = ry * (float)Hl + op[l * 8 + p * 2 + 1] - 0.5f;
            float x0f = floorf(x), y0f = floorf(y);
            int   x0 = (int)x0f, y0 = (int)y0f;
            float lx = x - x0f, ly = y - y0f;
            float w00 = (1.f - lx) * (1.f - ly), w10 = lx * (1.f - ly);
            float w01 = (1.f - lx) * ly,         w11 = lx * ly;
            float samp = 0.f;
            if (x0 >= 0     && x0 < Wl     && y0 >= 0     && y0 < Hl)
                samp += w00 * vbase[((size_t)y0 * Wl + x0) * 256];
            if (x0 + 1 >= 0 && x0 + 1 < Wl && y0 >= 0     && y0 < Hl)
                samp += w10 * vbase[((size_t)y0 * Wl + x0 + 1) * 256];
            if (x0 >= 0     && x0 < Wl     && y0 + 1 >= 0 && y0 + 1 < Hl)
                samp += w01 * vbase[((size_t)(y0 + 1) * Wl + x0) * 256];
            if (x0 + 1 >= 0 && x0 + 1 < Wl && y0 + 1 >= 0 && y0 + 1 < Hl)
                samp += w11 * vbase[((size_t)(y0 + 1) * Wl + x0 + 1) * 256];
            acc += e[l * 4 + p] * inv * samp;
        }
    }
    out[(size_t)bq * 256 + h * 32 + dh] = acc;
}

// ---------------------------------------------------------------------------
// Fused residual + LayerNorm over D=256; one wave per row.
// grid = ceil(M/8), block = 256.
// ---------------------------------------------------------------------------
__global__ __launch_bounds__(256) void ln_residual(
    const float* __restrict__ X, const float* __restrict__ R,
    const float* __restrict__ g, const float* __restrict__ bta,
    float* __restrict__ out, int M)
{
    const int lane = threadIdx.x & 31;
    const int row = blockIdx.x * 8 + (threadIdx.x >> 5);
    if (row >= M) return;
    const float* xp = X + (size_t)row * 256;
    const float* rp = R + (size_t)row * 256;
    float v[8];
    float sum = 0.f;
#pragma unroll
    for (int i = 0; i < 8; ++i) { v[i] = xp[lane + 32 * i] + rp[lane + 32 * i]; sum += v[i]; }
#pragma unroll
    for (int m = 1; m < 32; m <<= 1) sum += __shfl_xor(sum, m, 32);
    float mu = sum * (1.0f / 256.0f);
    float vs = 0.f;
#pragma unroll
    for (int i = 0; i < 8; ++i) { float d = v[i] - mu; vs += d * d; }
#pragma unroll
    for (int m = 1; m < 32; m <<= 1) vs += __shfl_xor(vs, m, 32);
    float rstd = rsqrtf(vs * (1.0f / 256.0f) + 1e-5f);
#pragma unroll
    for (int i = 0; i < 8; ++i) {
        int c = lane + 32 * i;
        out[(size_t)row * 256 + c] = (v[i] - mu) * rstd * g[c] + bta[c];
    }
}

__global__ void addvec(const float* __restrict__ a, const float* __restrict__ b,
                       float* __restrict__ c, int n)
{
    int i = blockIdx.x * blockDim.x + threadIdx.x;
    if (i < n) c[i] = a[i] + b[i];
}

// ---------------------------------------------------------------------------
// Host launcher
// ---------------------------------------------------------------------------
extern "C" void kernel_launch(void* const* d_in, const int* in_sizes, int n_in,
                              void* d_out, int out_size, void* d_ws, size_t ws_size,
                              hipStream_t stream) {
    (void)in_sizes; (void)n_in; (void)out_size; (void)ws_size;
    const float* tgt        = (const float*)d_in[0];
    const float* qpos       = (const float*)d_in[1];
    const float* refp       = (const float*)d_in[2];
    const float* src        = (const float*)d_in[3];
    const float* in_proj_w  = (const float*)d_in[6];
    const float* in_proj_b  = (const float*)d_in[7];
    const float* out_proj_w = (const float*)d_in[8];
    const float* out_proj_b = (const float*)d_in[9];
    const float* ln1_g      = (const float*)d_in[10];
    const float* ln1_b      = (const float*)d_in[11];
    const float* off_w      = (const float*)d_in[12];
    const float* off_b      = (const float*)d_in[13];
    const float* aw_w       = (const float*)d_in[14];
    const float* aw_b       = (const float*)d_in[15];
    const float* val_w      = (const float*)d_in[16];
    const float* val_b      = (const float*)d_in[17];
    const float* outp_w     = (const float*)d_in[18];
    const float* outp_b     = (const float*)d_in[19];
    const float* ln2_g      = (const float*)d_in[20];
    const float* ln2_b      = (const float*)d_in[21];
    const float* ff1_w      = (const float*)d_in[22];
    const float* ff1_b      = (const float*)d_in[23];
    const float* ff2_w      = (const float*)d_in[24];
    const float* ff2_b      = (const float*)d_in[25];
    const float* ln3_g      = (const float*)d_in[26];
    const float* ln3_b      = (const float*)d_in[27];

    float* ws = (float*)d_ws;
    const size_t NQD = (size_t)NQ_ * D_;
    float* qbuf = ws;
    float* Qp   = qbuf + NQD;
    float* Kp   = Qp   + NQD;
    float* Vp   = Kp   + NQD;
    float* Obuf = Vp   + NQD;
    float* attn = Obuf + NQD;
    float* tgt1 = attn + NQD;
    float* q2   = tgt1 + NQD;
    float* offb = q2   + NQD;
    float* awl  = offb + NQD;
    float* samp = awl  + (size_t)NQ_ * 128;
    float* t2   = samp + NQD;
    float* tgt2 = t2   + NQD;
    float* h1   = tgt2 + NQD;
    float* h2   = h1   + (size_t)NQ_ * DFF_;
    float* valb = h2   + NQD;

    auto gemm = [&](const float* A, const float* W, const float* bias, float* C,
                    int M, int N, int K, int relu) {
        dim3 g((M + 31) / 32, (N + 255) / 256);
        gemm_wmma_nt<<<g, dim3(256), 0, stream>>>(A, W, bias, C, M, N, K, relu);
    };

    const int n = NQ_ * D_;

    // q = tgt + query_pos
    addvec<<<(n + 255) / 256, 256, 0, stream>>>(tgt, qpos, qbuf, n);

    // self-attention projections (v uses tgt!)
    gemm(qbuf, in_proj_w,               in_proj_b,          Qp, NQ_, D_, D_, 0);
    gemm(qbuf, in_proj_w + D_ * D_,     in_proj_b + D_,     Kp, NQ_, D_, D_, 0);
    gemm(tgt,  in_proj_w + 2 * D_ * D_, in_proj_b + 2 * D_, Vp, NQ_, D_, D_, 0);

    attn_flash<<<dim3(B_ * H_, (Q_ + 15) / 16), 32, 0, stream>>>(Qp, Kp, Vp, Obuf);

    gemm(Obuf, out_proj_w, out_proj_b, attn, NQ_, D_, D_, 0);
    ln_residual<<<(NQ_ + 7) / 8, 256, 0, stream>>>(attn, tgt, ln1_g, ln1_b, tgt1, NQ_);

    // deformable attention branch
    addvec<<<(n + 255) / 256, 256, 0, stream>>>(tgt1, qpos, q2, n);
    gemm(src, val_w, val_b, valb, B_ * S_, D_, D_, 0);     // dominant GEMM
    gemm(q2,  off_w, off_b, offb, NQ_, D_,  D_, 0);
    gemm(q2,  aw_w,  aw_b,  awl,  NQ_, 128, D_, 0);
    msda_sample<<<(B_ * Q_ * H_ + 7) / 8, 256, 0, stream>>>(valb, offb, awl, refp, samp);
    gemm(samp, outp_w, outp_b, t2, NQ_, D_, D_, 0);
    ln_residual<<<(NQ_ + 7) / 8, 256, 0, stream>>>(t2, tgt1, ln2_g, ln2_b, tgt2, NQ_);

    // FFN
    gemm(tgt2, ff1_w, ff1_b, h1, NQ_, DFF_, D_, 1);
    gemm(h1,   ff2_w, ff2_b, h2, NQ_, D_, DFF_, 0);
    ln_residual<<<(NQ_ + 7) / 8, 256, 0, stream>>>(h2, tgt2, ln3_g, ln3_b, (float*)d_out, NQ_);
}
